// Model_45406394253668
// MI455X (gfx1250) — matrix operand order repaired and checked by reading, not compiled, not run on hardware
//
#include <hip/hip_runtime.h>
#include <hip/hip_bf16.h>
#include <math.h>

// ---------------- problem constants ----------------
#define BN   32
#define TYN  64
#define TXN  64
#define TTN  64
#define HD   512
#define VSN  32000
#define VTN  32000
#define NSPLIT 32
#define ROWS_EM  2064   // 2048 ts rows + 1 null row, padded to 16
#define ROWS_TG  2048
#define MT_EM    129    // ROWS_EM/16
#define MT_TG    128
#define EM_OUT   (BN*2*TYN*TXN)   // 262144

typedef __attribute__((ext_vector_type(16))) __bf16 v16bf;
typedef __attribute__((ext_vector_type(8)))  __bf16 v8bf;
typedef __attribute__((ext_vector_type(8)))  float  v8f;

// Build a v16bf WMMA fragment from two contiguous 8-element (16B) runs.
static __device__ __forceinline__ v16bf frag2(const __bf16* p0, const __bf16* p1) {
    v8bf lo = *(const v8bf*)p0;
    v8bf hi = *(const v8bf*)p1;
    v16bf r;
#pragma unroll
    for (int i = 0; i < 8; ++i) { r[i] = lo[i]; r[i + 8] = hi[i]; }
    return r;
}

// ---------------- conversion kernels ----------------
// y_hidden -> A0[0:2048], y_hidden_null -> A0[2048], zero-pad A0[2049:2064);
// target_state -> Atgt. All bf16, row-major [row][HD].
__global__ void build_inputs(const float* __restrict__ yh,
                             const float* __restrict__ ynull,
                             const float* __restrict__ tst,
                             __bf16* __restrict__ A0,
                             __bf16* __restrict__ Atgt) {
    size_t i = (size_t)blockIdx.x * 256 + threadIdx.x;
    if (i < (size_t)2048 * HD) {
        A0[i]   = (__bf16)yh[i];
        Atgt[i] = (__bf16)tst[i];
    } else if (i < (size_t)2049 * HD) {
        A0[i] = (__bf16)ynull[i - (size_t)2048 * HD];
    } else if (i < (size_t)ROWS_EM * HD) {
        A0[i] = (__bf16)0.0f;
    }
}

// Transpose-convert W[K][N] fp32 -> WT[N][K] bf16 via LDS 32x32 tiles.
__global__ void transpose_to_bf16(const float* __restrict__ in,
                                  __bf16* __restrict__ out, int K, int N) {
    __shared__ float tile[32][33];
    int n0 = blockIdx.x * 32, k0 = blockIdx.y * 32;
    for (int i = threadIdx.y; i < 32; i += 8) {
        int k = k0 + i, n = n0 + threadIdx.x;
        tile[i][threadIdx.x] = (k < K && n < N) ? in[(size_t)k * N + n] : 0.0f;
    }
    __syncthreads();
    for (int i = threadIdx.y; i < 32; i += 8) {
        int n = n0 + i, k = k0 + threadIdx.x;
        if (n < N && k < K) out[(size_t)n * K + k] = (__bf16)tile[threadIdx.x][i];
    }
}

// ---------------- tanh GEMM: TS = tanh(A0 @ WemT^T + b_em) in bf16 ----------------
__global__ __launch_bounds__(32)
void tanh_gemm(const __bf16* __restrict__ A0,
               const __bf16* __restrict__ WemT,   // [HD][HD] (N-major)
               const float*  __restrict__ bem,
               __bf16* __restrict__ TS) {
    __shared__ __attribute__((aligned(16))) __bf16 atile[16 * HD];
    const int lane = threadIdx.x, mtile = blockIdx.x, split = blockIdx.y;
    {   // stage 16x512 bf16 A tile (contiguous 16KB)
        const v8bf* s = (const v8bf*)(A0 + (size_t)mtile * 16 * HD);
        v8bf* d = (v8bf*)atile;
        for (int i = lane; i < 16 * HD / 8; i += 32) d[i] = s[i];
    }
    const int mrow = lane & 15;
    const int kA0 = (lane < 16) ? 0 : 8;
    const int kB0 = (lane < 16) ? 0 : 8;
    const int off = (lane < 16) ? 0 : 8;
    for (int nt = split * 4; nt < split * 4 + 4; ++nt) {   // 8 splits x 4 tiles = 32
        const int n = nt * 16 + (lane & 15);
        const __bf16* wcol = WemT + (size_t)n * HD;
        v8f c = {};
#pragma unroll
        for (int kk = 0; kk < HD / 32; ++kk) {
            v16bf a = frag2(atile + mrow * HD + kk * 32 + kA0,
                            atile + mrow * HD + kk * 32 + kA0 + 16);
            v16bf b = frag2(wcol + kk * 32 + kB0, wcol + kk * 32 + kB0 + 16);
            c = __builtin_amdgcn_wmma_f32_16x16x32_bf16(false, a, false, b,
                                                        (short)0, c, false, false);
        }
        const float bb = bem[n];
#pragma unroll
        for (int r = 0; r < 8; ++r) {
            const int row = mtile * 16 + off + r;
            TS[(size_t)row * HD + n] = (__bf16)tanhf(c[r] + bb);
        }
    }
}

// -------- fused GEMM + online softmax statistics (sumexp, max, argmax) --------
__global__ __launch_bounds__(32)
void gemm_softmax_stats(const __bf16* __restrict__ A,    // [rows][HD] bf16
                        const __bf16* __restrict__ WT,   // [V][HD] bf16
                        const float*  __restrict__ bias, // [V]
                        float* __restrict__ psum,
                        float* __restrict__ pmax,
                        int*   __restrict__ pidx,
                        int vtiles) {
    __shared__ __attribute__((aligned(16))) __bf16 atile[16 * HD];
    const int lane = threadIdx.x, mtile = blockIdx.x, split = blockIdx.y;
    const int nsplit = gridDim.y;
    {
        const v8bf* s = (const v8bf*)(A + (size_t)mtile * 16 * HD);
        v8bf* d = (v8bf*)atile;
        for (int i = lane; i < 16 * HD / 8; i += 32) d[i] = s[i];
    }
    const int tp = (vtiles + nsplit - 1) / nsplit;
    const int t0 = split * tp;
    int t1 = t0 + tp; if (t1 > vtiles) t1 = vtiles;

    float sacc[8]; float vmax[8]; int vidx[8];
#pragma unroll
    for (int r = 0; r < 8; ++r) { sacc[r] = 0.0f; vmax[r] = -3.4e38f; vidx[r] = 0; }

    const int mrow = lane & 15;
    const int kA0 = (lane < 16) ? 0 : 8;
    const int kB0 = (lane < 16) ? 0 : 8;

    for (int nt = t0; nt < t1; ++nt) {
        const int n = nt * 16 + (lane & 15);
        const __bf16* wcol = WT + (size_t)n * HD;
        __builtin_prefetch(wcol + HD * 16, 0, 1);   // next N-tile column stream
        v8f c = {};
#pragma unroll
        for (int kk = 0; kk < HD / 32; ++kk) {      // 16 back-to-back WMMAs
            v16bf a = frag2(atile + mrow * HD + kk * 32 + kA0,
                            atile + mrow * HD + kk * 32 + kA0 + 16);
            v16bf b = frag2(wcol + kk * 32 + kB0, wcol + kk * 32 + kB0 + 16);
            c = __builtin_amdgcn_wmma_f32_16x16x32_bf16(false, a, false, b,
                                                        (short)0, c, false, false);
        }
        const float bb = bias[n];
#pragma unroll
        for (int r = 0; r < 8; ++r) {               // online stats, no materialization
            float v = c[r] + bb;
            sacc[r] += __expf(v);
            if (v > vmax[r]) { vmax[r] = v; vidx[r] = n; }
        }
    }
    // butterfly reduce across the 16-lane halves (rows r / 8+r)
#pragma unroll
    for (int m = 1; m < 16; m <<= 1) {
#pragma unroll
        for (int r = 0; r < 8; ++r) {
            sacc[r] += __shfl_xor(sacc[r], m, 32);
            float om = __shfl_xor(vmax[r], m, 32);
            int   oi = __shfl_xor(vidx[r], m, 32);
            if (om > vmax[r] || (om == vmax[r] && oi < vidx[r])) { vmax[r] = om; vidx[r] = oi; }
        }
    }
    if ((lane & 15) == 0) {
        const int off = (lane >> 4) * 8;
#pragma unroll
        for (int r = 0; r < 8; ++r) {
            const size_t row = (size_t)(mtile * 16 + off + r);
            psum[row * nsplit + split] = sacc[r];
            pmax[row * nsplit + split] = vmax[r];
            pidx[row * nsplit + split] = vidx[r];
        }
    }
}

__global__ __launch_bounds__(32)
void finalize_stats(const float* __restrict__ psum, const float* __restrict__ pmax,
                    const int* __restrict__ pidx,
                    float* __restrict__ sum, int* __restrict__ amax) {
    const int row = blockIdx.x, lane = threadIdx.x;   // NSPLIT == warpSize == 32
    float s = psum[(size_t)row * NSPLIT + lane];
    float m = pmax[(size_t)row * NSPLIT + lane];
    int   i = pidx[(size_t)row * NSPLIT + lane];
#pragma unroll
    for (int d = 1; d < 32; d <<= 1) {
        s += __shfl_xor(s, d, 32);
        float om = __shfl_xor(m, d, 32);
        int   oi = __shfl_xor(i, d, 32);
        if (om > m || (om == m && oi < i)) { m = om; i = oi; }
    }
    if (lane == 0) { sum[row] = s; amax[row] = i; }
}

// -------- emission gather-GEMM: em_g[b,ty,tx] = exp(logit at sources)/sumexp --------
__global__ __launch_bounds__(32)
void emission_gather(const __bf16* __restrict__ TS,
                     const __bf16* __restrict__ WsvT,
                     const float*  __restrict__ bsv,
                     const int*    __restrict__ sources,
                     const float*  __restrict__ sumexp,
                     float* __restrict__ out) {
    __shared__ __attribute__((aligned(16))) __bf16 atile[16 * HD];
    const int lane = threadIdx.x, b = blockIdx.x, mtile = blockIdx.y;
    {
        const v8bf* s = (const v8bf*)(TS + ((size_t)b * TYN + mtile * 16) * HD);
        v8bf* d = (v8bf*)atile;
        for (int i = lane; i < 16 * HD / 8; i += 32) d[i] = s[i];
    }
    const int mrow = lane & 15;
    const int kA0 = (lane < 16) ? 0 : 8;
    const int kB0 = (lane < 16) ? 0 : 8;
    const int off = (lane < 16) ? 0 : 8;
    for (int nt = 0; nt < TXN / 16; ++nt) {
        const int tx = nt * 16 + (lane & 15);
        const int v = sources[b * TXN + tx];
        const __bf16* wcol = WsvT + (size_t)v * HD;
        v8f c = {};
#pragma unroll
        for (int kk = 0; kk < HD / 32; ++kk) {
            v16bf a = frag2(atile + mrow * HD + kk * 32 + kA0,
                            atile + mrow * HD + kk * 32 + kA0 + 16);
            v16bf bfr = frag2(wcol + kk * 32 + kB0, wcol + kk * 32 + kB0 + 16);
            c = __builtin_amdgcn_wmma_f32_16x16x32_bf16(false, a, false, bfr,
                                                        (short)0, c, false, false);
        }
        const float bb = bsv[v];
#pragma unroll
        for (int r = 0; r < 8; ++r) {
            const int ty = mtile * 16 + off + r;
            const float se = sumexp[b * TYN + ty];
            out[(size_t)b * (2 * TYN * TXN) + (size_t)ty * TXN + tx] =
                __expf(c[r] + bb) / se;
        }
    }
}

// null emission row: one wave per (b, tx); broadcast over all 64 ty rows
__global__ __launch_bounds__(32)
void emission_null(const __bf16* __restrict__ TS,
                   const __bf16* __restrict__ WsvT,
                   const float*  __restrict__ bsv,
                   const int*    __restrict__ sources,
                   const float*  __restrict__ sumexp,
                   float* __restrict__ out) {
    const int lane = threadIdx.x;
    const int b = blockIdx.x >> 6, tx = blockIdx.x & 63;
    const int v = sources[b * TXN + tx];
    const __bf16* arow = TS + (size_t)2048 * HD;     // tsn row
    const __bf16* wcol = WsvT + (size_t)v * HD;
    const int k0 = lane * 16;
    float acc = 0.0f;
#pragma unroll
    for (int i = 0; i < 16; ++i) acc += (float)arow[k0 + i] * (float)wcol[k0 + i];
#pragma unroll
    for (int m = 1; m < 32; m <<= 1) acc += __shfl_xor(acc, m, 32);
    const float prob = __expf(acc + bsv[v]) / sumexp[2048];
    out[(size_t)b * (2 * TYN * TXN) + (size_t)(TYN + lane) * TXN + tx] = prob;
    out[(size_t)b * (2 * TYN * TXN) + (size_t)(TYN + 32 + lane) * TXN + tx] = prob;
}

// reconstruction: prob of own token, its log, mask, argmax; one wave per (b,t)
__global__ __launch_bounds__(32)
void reconstruction(const __bf16* __restrict__ Atgt,
                    const __bf16* __restrict__ WtvT,
                    const float*  __restrict__ btv,
                    const int*    __restrict__ targets,
                    const int*    __restrict__ tlen,
                    const float*  __restrict__ sumexp,
                    const int*    __restrict__ amax,
                    float* __restrict__ out) {
    const int lane = threadIdx.x, row = blockIdx.x;
    const int b = row >> 6, t = row & 63;
    const int v = targets[row];
    const __bf16* arow = Atgt + (size_t)row * HD;
    const __bf16* wcol = WtvT + (size_t)v * HD;
    const int k0 = lane * 16;
    float acc = 0.0f;
#pragma unroll
    for (int i = 0; i < 16; ++i) acc += (float)arow[k0 + i] * (float)wcol[k0 + i];
#pragma unroll
    for (int m = 1; m < 32; m <<= 1) acc += __shfl_xor(acc, m, 32);
    if (lane == 0) {
        const float logit = acc + btv[v];
        const float se = sumexp[row];
        const float mask = (t < tlen[b]) ? 1.0f : 0.0f;
        out[EM_OUT + row]            = (__expf(logit) / se) * mask;
        out[EM_OUT + 2048 + row]     = (logit - __logf(se)) * mask;
        out[EM_OUT + 4096 + row]     = (float)amax[row];
    }
}

// ---------------- launch ----------------
extern "C" void kernel_launch(void* const* d_in, const int* in_sizes, int n_in,
                              void* d_out, int out_size, void* d_ws, size_t ws_size,
                              hipStream_t stream) {
    const float* yh    = (const float*)d_in[0];
    const float* ynull = (const float*)d_in[1];
    const float* tst   = (const float*)d_in[2];
    const float* Wem   = (const float*)d_in[3];
    const float* bem   = (const float*)d_in[4];
    const float* Wsv   = (const float*)d_in[5];
    const float* bsv   = (const float*)d_in[6];
    const float* Wtv   = (const float*)d_in[7];
    const float* btv   = (const float*)d_in[8];
    const int* sources = (const int*)d_in[9];
    const int* targets = (const int*)d_in[10];
    const int* tlen    = (const int*)d_in[11];
    float* out = (float*)d_out;

    char* ws = (char*)d_ws;
    size_t off = 0;
    auto alloc = [&](size_t bytes) -> void* {
        off = (off + 255) & ~(size_t)255;
        void* p = ws + off; off += bytes; return p;
    };
    __bf16* A0   = (__bf16*)alloc((size_t)ROWS_EM * HD * 2);
    __bf16* TS   = (__bf16*)alloc((size_t)ROWS_EM * HD * 2);
    __bf16* Atgt = (__bf16*)alloc((size_t)ROWS_TG * HD * 2);
    __bf16* WemT = (__bf16*)alloc((size_t)HD * HD * 2);
    __bf16* WsvT = (__bf16*)alloc((size_t)VSN * HD * 2);
    __bf16* WtvT = (__bf16*)alloc((size_t)VTN * HD * 2);
    float* emPS = (float*)alloc((size_t)ROWS_EM * NSPLIT * 4);
    float* emPM = (float*)alloc((size_t)ROWS_EM * NSPLIT * 4);
    int*   emPI = (int*)  alloc((size_t)ROWS_EM * NSPLIT * 4);
    float* tgPS = (float*)alloc((size_t)ROWS_TG * NSPLIT * 4);
    float* tgPM = (float*)alloc((size_t)ROWS_TG * NSPLIT * 4);
    int*   tgPI = (int*)  alloc((size_t)ROWS_TG * NSPLIT * 4);
    float* emSum = (float*)alloc((size_t)ROWS_EM * 4);
    int*   emIdx = (int*)  alloc((size_t)ROWS_EM * 4);
    float* tgSum = (float*)alloc((size_t)ROWS_TG * 4);
    int*   tgIdx = (int*)  alloc((size_t)ROWS_TG * 4);

    build_inputs<<<(ROWS_EM * HD) / 256, 256, 0, stream>>>(yh, ynull, tst, A0, Atgt);
    transpose_to_bf16<<<dim3(HD / 32, HD / 32), dim3(32, 8), 0, stream>>>(Wem, WemT, HD, HD);
    transpose_to_bf16<<<dim3(VSN / 32, HD / 32), dim3(32, 8), 0, stream>>>(Wsv, WsvT, HD, VSN);
    transpose_to_bf16<<<dim3(VTN / 32, HD / 32), dim3(32, 8), 0, stream>>>(Wtv, WtvT, HD, VTN);

    tanh_gemm<<<dim3(MT_EM, 8), 32, 0, stream>>>(A0, WemT, bem, TS);

    gemm_softmax_stats<<<dim3(MT_EM, NSPLIT), 32, 0, stream>>>(TS, WsvT, bsv,
                                                               emPS, emPM, emPI, VSN / 16);
    gemm_softmax_stats<<<dim3(MT_TG, NSPLIT), 32, 0, stream>>>(Atgt, WtvT, btv,
                                                               tgPS, tgPM, tgPI, VTN / 16);
    finalize_stats<<<ROWS_EM, 32, 0, stream>>>(emPS, emPM, emPI, emSum, emIdx);
    finalize_stats<<<ROWS_TG, 32, 0, stream>>>(tgPS, tgPM, tgPI, tgSum, tgIdx);

    emission_gather<<<dim3(BN, TYN / 16), 32, 0, stream>>>(TS, WsvT, bsv, sources, emSum, out);
    emission_null<<<BN * TXN, 32, 0, stream>>>(TS, WsvT, bsv, sources, emSum, out);
    reconstruction<<<BN * TTN, 32, 0, stream>>>(Atgt, WtvT, btv, targets, tlen,
                                                tgSum, tgIdx, out);
}
